// biLSTM_72550587564391
// MI455X (gfx1250) — matrix operand order repaired and checked by reading, not compiled, not run on hardware
//
#include <hip/hip_runtime.h>
#include <math.h>

// ---------------------------------------------------------------------------
// Bi-LSTM for MI455X (gfx1250, wave32, WMMA bf16 16x16x32, f32 accumulate).
//  Phase 1: repack weights to bf16 (Wa [2][2048][512], Wx [2][2048][320]),
//           stack biases, pad W_p to [16][1024], zero a double-buffers.
//  Phase 2: embedding gather -> Xe bf16 [T*B][320] (K padded 300->320).
//  Phase 3: input projection Zx = Xe @ Wx^T  (big parallel WMMA GEMM).
//  Phase 4: persistent recurrence: 32 WGs (2 dirs x 16 h-slices), Wa slice
//           staged into LDS once via TDM tensor_load_to_lds (hardware row
//           padding), resident for all 256 steps; cell state in registers;
//           device-wide atomic barrier + s_sleep per step; global_prefetch
//           of next step's Zx slice overlapped with the WMMA k-loop.
//  Phase 5: projection a_seq[16384][1024] @ W_p^T via WMMA (N padded to 16).
// ---------------------------------------------------------------------------

typedef __attribute__((ext_vector_type(16))) __bf16 bf16x16;
typedef __attribute__((ext_vector_type(8)))  float  floatx8;
typedef __attribute__((ext_vector_type(4)))  unsigned u32x4;
typedef __attribute__((ext_vector_type(8)))  unsigned u32x8;

#define T_SEQ   256
#define B_SZ    64
#define E_DIM   300
#define EPAD    320
#define H_DIM   512
#define O_DIM   9
#define C4H     2048
#define NTB     16384            // T_SEQ * B_SZ
#define LDSK    520              // padded K stride (bf16 elems) for LDS Wa tile
#define NWG_REC 32               // 2 dirs * 16 slices

union FragBF { bf16x16 v; uint4 q[2]; };

__device__ __forceinline__ floatx8 wmma_bf16(bf16x16 a, bf16x16 b, floatx8 c) {
  return __builtin_amdgcn_wmma_f32_16x16x32_bf16(false, a, false, b, (short)0, c,
                                                 false, false);
}

// A-fragment (16x32, MxK): lane l -> row m = l&15, k-base = (l>>4)*16;
// elements 0..15 = K[kb..kb+15].
__device__ __forceinline__ bf16x16 load_fragA(const __bf16* p, int ldk, int lane) {
  FragBF f;
  const __bf16* q = p + (size_t)(lane & 15) * ldk + ((lane >> 4) << 4);
  f.q[0] = *(const uint4*)(q);
  f.q[1] = *(const uint4*)(q + 8);
  return f.v;
}

// B-fragment (32x16, KxN) from row-major [N][K] weights: lane l -> col n = l&15,
// 16 consecutive K starting at (l>>4)*16 -> 32 contiguous bytes.
__device__ __forceinline__ bf16x16 load_fragB(const __bf16* p, int ldk, int lane) {
  FragBF f;
  const __bf16* q = p + (size_t)(lane & 15) * ldk + ((lane >> 4) << 4);
  f.q[0] = *(const uint4*)(q);
  f.q[1] = *(const uint4*)(q + 8);
  return f.v;
}

__device__ __forceinline__ float sigmoidf_(float x) { return 1.f / (1.f + __expf(-x)); }

// -------------------- workspace layout (byte offsets, 256-aligned) ----------
static constexpr size_t SZ_XE  = (size_t)NTB * EPAD * 2;
static constexpr size_t SZ_WA  = (size_t)2 * C4H * H_DIM * 2;
static constexpr size_t SZ_WX  = (size_t)2 * C4H * EPAD * 2;
static constexpr size_t SZ_BG  = (size_t)2 * C4H * 4;
static constexpr size_t SZ_WP  = (size_t)16 * 1024 * 2;
static constexpr size_t SZ_AB  = (size_t)2 * 2 * B_SZ * H_DIM * 2;
static constexpr size_t SZ_ZX  = (size_t)2 * NTB * C4H * 2;
static constexpr size_t SZ_ASQ = (size_t)NTB * 1024 * 2;

static constexpr size_t OFF_XE  = 0;
static constexpr size_t OFF_WA  = OFF_XE + SZ_XE;
static constexpr size_t OFF_WX  = OFF_WA + SZ_WA;
static constexpr size_t OFF_BG  = OFF_WX + SZ_WX;
static constexpr size_t OFF_WP  = OFF_BG + SZ_BG;
static constexpr size_t OFF_AB  = OFF_WP + SZ_WP;
static constexpr size_t OFF_BAR = OFF_AB + SZ_AB;
static constexpr size_t OFF_ZX  = OFF_BAR + 256;
static constexpr size_t OFF_ASQ = OFF_ZX + SZ_ZX;

// -------------------- device-wide barrier ----------------------------------
__device__ __forceinline__ void grid_sync(unsigned* bar, unsigned target) {
  __syncthreads();
  if (threadIdx.x == 0) {
    __hip_atomic_fetch_add(bar, 1u, __ATOMIC_RELEASE, __HIP_MEMORY_SCOPE_AGENT);
    while (__hip_atomic_load(bar, __ATOMIC_ACQUIRE, __HIP_MEMORY_SCOPE_AGENT) < target) {
      __builtin_amdgcn_s_sleep(2);
    }
  }
  __syncthreads();
}

// -------------------- phase 1a: weight repack ------------------------------
struct WPtrs { const float* W[8]; const float* b[8]; };  // [dir*4 + gate], order f,i,o,c

__global__ void prep_weights_kernel(WPtrs p, __bf16* __restrict__ Wa,
                                    __bf16* __restrict__ Wx, float* __restrict__ bg) {
  const int rg  = blockIdx.x;          // 0..4095 = dir*2048 + r
  const int dir = rg >> 11;
  const int r   = rg & 2047;
  const int g   = r >> 9;
  const int rr  = r & 511;
  const float* W = p.W[dir * 4 + g];   // [512][812], cols 0..511 = a-part, 512..811 = x-part
  for (int k = threadIdx.x; k < H_DIM + EPAD; k += blockDim.x) {
    if (k < H_DIM) {
      Wa[((size_t)dir * C4H + r) * H_DIM + k] = (__bf16)W[rr * 812 + k];
    } else {
      int j = k - H_DIM;
      Wx[((size_t)dir * C4H + r) * EPAD + j] =
          (j < E_DIM) ? (__bf16)W[rr * 812 + H_DIM + j] : (__bf16)0.f;
    }
  }
  if (threadIdx.x == 0) bg[dir * C4H + r] = p.b[dir * 4 + g][rr];
}

// -------------------- phase 1b: W_p pad, a-buffers, barrier ----------------
__global__ void prep_misc_kernel(const float* __restrict__ Wp_src, __bf16* __restrict__ Wp,
                                 __bf16* __restrict__ abuf, unsigned* __restrict__ bar) {
  const int tid = blockIdx.x * blockDim.x + threadIdx.x;   // 64*256 = 16384
  const int o = tid >> 10, k = tid & 1023;
  Wp[tid] = (o < O_DIM) ? (__bf16)Wp_src[o * 1024 + k] : (__bf16)0.f;
  for (int i = tid; i < 2 * 2 * B_SZ * H_DIM; i += 16384) abuf[i] = (__bf16)0.f;
  if (tid == 0) bar[0] = 0u;
}

// -------------------- phase 2: embedding gather ----------------------------
__global__ void embed_kernel(const int* __restrict__ x, const float* __restrict__ emb,
                             __bf16* __restrict__ Xe) {
  const int r = blockIdx.x;            // 0..16383 = t*64 + b
  const int t = r >> 6, b = r & 63;
  const int idx = x[b * T_SEQ + t];
  const float* src = emb + (size_t)idx * E_DIM;
  __bf16* dst = Xe + (size_t)r * EPAD;
  for (int e = threadIdx.x; e < EPAD; e += blockDim.x)
    dst[e] = (e < E_DIM) ? (__bf16)src[e] : (__bf16)0.f;
}

// -------------------- phase 3: Zx = Xe @ Wx^T (per direction) --------------
__global__ void zx_gemm_kernel(const __bf16* __restrict__ Xe,   // [16384][320]
                               const __bf16* __restrict__ Wx,   // [2][2048][320]
                               __bf16* __restrict__ Zx) {       // [2][16384][2048]
  const int dir   = blockIdx.z;
  const int m0    = blockIdx.x * 64;
  const int n0    = blockIdx.y * 256;
  const int lane  = threadIdx.x & 31;
  const int wave  = threadIdx.x >> 5;
  const int wm    = wave & 1;
  const int wn    = wave >> 1;
  const int lrow  = lane & 15;
  const int lhalf = lane >> 4;
  const int mw    = m0 + wm * 32;
  const int nw    = n0 + wn * 64;
  const __bf16* Wxd = Wx + (size_t)dir * C4H * EPAD;

  floatx8 acc[2][4];
#pragma unroll
  for (int mt = 0; mt < 2; ++mt)
#pragma unroll
    for (int nt = 0; nt < 4; ++nt)
      acc[mt][nt] = floatx8{0.f, 0.f, 0.f, 0.f, 0.f, 0.f, 0.f, 0.f};

  for (int ks = 0; ks < EPAD / 32; ++ks) {
    const int k0 = ks << 5;
    bf16x16 af[2];
#pragma unroll
    for (int mt = 0; mt < 2; ++mt)
      af[mt] = load_fragA(Xe + (size_t)(mw + mt * 16) * EPAD + k0, EPAD, lane);
#pragma unroll
    for (int nt = 0; nt < 4; ++nt) {
      bf16x16 bf = load_fragB(Wxd + (size_t)(nw + nt * 16) * EPAD + k0, EPAD, lane);
#pragma unroll
      for (int mt = 0; mt < 2; ++mt)
        acc[mt][nt] = wmma_bf16(af[mt], bf, acc[mt][nt]);
    }
  }

  __bf16* out = Zx + (size_t)dir * NTB * C4H;
#pragma unroll
  for (int mt = 0; mt < 2; ++mt)
#pragma unroll
    for (int nt = 0; nt < 4; ++nt)
#pragma unroll
      for (int v = 0; v < 8; ++v) {
        const int m = mw + mt * 16 + v + lhalf * 8;
        const int n = nw + nt * 16 + lrow;
        out[(size_t)m * C4H + n] = (__bf16)acc[mt][nt][v];
      }
}

// -------------------- phase 4: persistent recurrence -----------------------
// 32 WGs: dir = wg>>4, slice s = wg&15 (h in [s*32, s*32+32)).
// Wa slice = 3D tile (K=512 x 32 rows x 4 gate blocks) DMA'd into LDS by the
// Tensor Data Mover with hardware padding (512 elems -> LDSK=520 row stride).
__global__ void lstm_recurrent_kernel(const __bf16* __restrict__ Wa,   // [2][2048][512]
                                      const __bf16* __restrict__ Zx,   // [2][16384][2048]
                                      const float*  __restrict__ bg,   // [2][2048]
                                      __bf16* __restrict__ abuf,       // [2][2][64][512]
                                      __bf16* __restrict__ aseq,       // [16384][1024]
                                      unsigned* __restrict__ bar) {
  extern __shared__ __bf16 ldsW[];     // [128][LDSK]
  const int wg    = blockIdx.x;
  const int dir   = wg >> 4;
  const int s     = wg & 15;
  const int tid   = threadIdx.x;
  const int lane  = tid & 31;
  const int wave  = tid >> 5;
  const int wm    = wave & 3;          // M tile (16 batch rows)
  const int wn    = wave >> 2;         // h 16-col slice (0..1)
  const int lrow  = lane & 15;
  const int lhalf = lane >> 4;

  // ---- TDM stage: tensor_load_to_lds of the Wa slice (wave 0 issues) ----
  // Tile: dim0 = 512 elems (2B), dim1 = 32 rows (stride 512), dim2 = 4 gate
  // blocks (stride 512*512). LDS pad: every 256 DWORDs (one row) add 4 DWORDs.
  if (tid < 32) {
    const unsigned long long gaddr =
        (unsigned long long)(uintptr_t)(Wa + (size_t)dir * C4H * H_DIM +
                                        (size_t)(s * 32) * H_DIM);
    const unsigned ldsb = __builtin_amdgcn_groupstaticsize();  // dynamic-LDS base
    u32x4 g0;
    g0[0] = 1u;                                        // count=1, user D#
    g0[1] = ldsb;                                      // lds_addr
    g0[2] = (unsigned)(gaddr & 0xffffffffull);         // global_addr[31:0]
    g0[3] = (unsigned)((gaddr >> 32) & 0x01ffffffull)  // global_addr[56:32]
          | 0x80000000u;                               // type = 2 ("image")
    u32x8 g1;
    g1[0] = (1u << 16)        // data_size = 2 bytes
          | (1u << 20)        // pad_enable
          | (7u << 22)        // pad_interval = 256 DWORDs (one 1024B row)
          | (3u << 25);       // pad_amount   = 4 DWORDs (8 bf16 -> LDSK=520)
    g1[1] = (unsigned)(H_DIM) << 16;                   // tensor_dim0[15:0] = 512
    g1[2] = (unsigned)(C4H) << 16;                     // tensor_dim1[15:0] = 2048
    g1[3] = (unsigned)(H_DIM) << 16;                   // tile_dim0 = 512
    g1[4] = 32u | (4u << 16);                          // tile_dim1 = 32, tile_dim2 = 4
    g1[5] = (unsigned)H_DIM;                           // tensor_dim0_stride = 512
    g1[6] = 0u;                                        // dim1_stride[15:0] = 0
    g1[7] = 4u;                                        // dim1_stride[47:16]: 262144
    u32x4 g2;
    g2[0] = 4u;                                        // tensor_dim2 = 4
    g2[1] = 1u;                                        // tensor_dim3 = 1
    g2[2] = 0x100000u;                                 // tensor_dim2_stride = 1M elems
    g2[3] = 0u;                                        // tile_dim3 unused
    u32x4 g3;
    g3[0] = 0u;
    g3[1] = 1u << 16;                                  // tensor_dim4 = 1
    g3[2] = 0u;
    g3[3] = 0u;
    asm volatile("tensor_load_to_lds %0, %1, %2, %3"
                 :: "s"(g0), "s"(g1), "s"(g2), "s"(g3)
                 : "memory");
    __builtin_amdgcn_s_wait_tensorcnt(0);
  }
  __syncthreads();

  const int hcol = (s << 5) + (wn << 4) + lrow;        // h in [0,512)
  float bgv[4];
#pragma unroll
  for (int g = 0; g < 4; ++g) bgv[g] = bg[dir * C4H + (g << 9) + hcol];

  float creg[8];
#pragma unroll
  for (int v = 0; v < 8; ++v) creg[v] = 0.f;

  const __bf16* Zxd = Zx + (size_t)dir * NTB * C4H;

  for (int it = 0; it < T_SEQ; ++it) {
    const int tt = dir ? (T_SEQ - 1 - it) : it;
    const int rb = it & 1, wb = rb ^ 1;
    const __bf16* aprev = abuf + ((size_t)dir * 2 + rb) * B_SZ * H_DIM;
    __bf16*       anew  = abuf + ((size_t)dir * 2 + wb) * B_SZ * H_DIM;

    // Prefetch next step's Zx slice (hidden behind the WMMA k-loop).
    if (it + 1 < T_SEQ) {
      const int tn = dir ? (T_SEQ - 2 - it) : (it + 1);
      const __bf16* pz = Zxd + (size_t)(tn * B_SZ + (wm << 4) + lrow) * C4H
                       + (s << 5) + (wn << 4);
#pragma unroll
      for (int g = 0; g < 4; ++g)
        __builtin_prefetch((const void*)(pz + (g << 9)), 0, 1);
    }

    floatx8 acc[4];
#pragma unroll
    for (int g = 0; g < 4; ++g)
      acc[g] = floatx8{0.f, 0.f, 0.f, 0.f, 0.f, 0.f, 0.f, 0.f};

#pragma unroll 4
    for (int ks = 0; ks < H_DIM / 32; ++ks) {
      const int k0 = ks << 5;
      bf16x16 af = load_fragA(aprev + (size_t)(wm << 4) * H_DIM + k0, H_DIM, lane);
#pragma unroll
      for (int g = 0; g < 4; ++g) {
        FragBF bf;
        const int off = ((g << 5) + (wn << 4) + lrow) * LDSK + k0 + (lhalf << 4);
        bf.q[0] = *(const uint4*)&ldsW[off];
        bf.q[1] = *(const uint4*)&ldsW[off + 8];
        acc[g] = wmma_bf16(af, bf.v, acc[g]);
      }
    }

    // gate nonlinearity + state update (all operands for (b,h) in this wave)
#pragma unroll
    for (int v = 0; v < 8; ++v) {
      const int b = (wm << 4) + v + (lhalf << 3);
      const size_t zrow = (size_t)(tt * B_SZ + b) * C4H;
      const float zf = acc[0][v] + (float)Zxd[zrow + 0 * H_DIM + hcol] + bgv[0];
      const float zi = acc[1][v] + (float)Zxd[zrow + 1 * H_DIM + hcol] + bgv[1];
      const float zo = acc[2][v] + (float)Zxd[zrow + 2 * H_DIM + hcol] + bgv[2];
      const float zc = acc[3][v] + (float)Zxd[zrow + 3 * H_DIM + hcol] + bgv[3];
      const float c  = sigmoidf_(zf) * creg[v] + sigmoidf_(zi) * tanhf(zc);
      creg[v] = c;
      const float a = sigmoidf_(zo) * tanhf(c);
      const __bf16 ab = (__bf16)a;
      anew[(size_t)b * H_DIM + hcol] = ab;
      aseq[(size_t)(tt * B_SZ + b) * 1024 + dir * H_DIM + hcol] = ab;
    }

    grid_sync(bar, (unsigned)(NWG_REC * (it + 1)));
  }
}

// -------------------- phase 5: output projection ---------------------------
__global__ void proj_kernel(const __bf16* __restrict__ aseq,   // [16384][1024]
                            const __bf16* __restrict__ Wp,     // [16][1024]
                            const float* __restrict__ bp,      // [9]
                            float* __restrict__ out) {         // [64][256][9]
  const int lane  = threadIdx.x & 31;
  const int wave  = threadIdx.x >> 5;
  const int tile  = blockIdx.x * 8 + wave;     // 1024 tiles of 16 rows
  const int m0    = tile * 16;
  const int lrow  = lane & 15;
  const int lhalf = lane >> 4;

  floatx8 acc = floatx8{0.f, 0.f, 0.f, 0.f, 0.f, 0.f, 0.f, 0.f};
#pragma unroll 8
  for (int ks = 0; ks < 1024 / 32; ++ks) {
    const int k0 = ks << 5;
    bf16x16 a = load_fragA(aseq + (size_t)m0 * 1024 + k0, 1024, lane);
    bf16x16 b = load_fragB(Wp + k0, 1024, lane);
    acc = wmma_bf16(a, b, acc);
  }
  if (lrow < O_DIM) {
    const float bias = bp[lrow];
#pragma unroll
    for (int v = 0; v < 8; ++v) {
      const int m = m0 + v + lhalf * 8;        // m = t*64 + b
      const int t = m >> 6, b = m & 63;
      out[((size_t)b * T_SEQ + t) * O_DIM + lrow] = acc[v] + bias;
    }
  }
}

// ---------------------------------------------------------------------------
extern "C" void kernel_launch(void* const* d_in, const int* in_sizes, int n_in,
                              void* d_out, int out_size, void* d_ws, size_t ws_size,
                              hipStream_t stream) {
  (void)in_sizes; (void)n_in; (void)out_size; (void)ws_size;
  char* ws = (char*)d_ws;
  __bf16*   Xe   = (__bf16*)(ws + OFF_XE);
  __bf16*   Wa   = (__bf16*)(ws + OFF_WA);
  __bf16*   Wx   = (__bf16*)(ws + OFF_WX);
  float*    bg   = (float*)(ws + OFF_BG);
  __bf16*   Wp   = (__bf16*)(ws + OFF_WP);
  __bf16*   ab   = (__bf16*)(ws + OFF_AB);
  unsigned* bar  = (unsigned*)(ws + OFF_BAR);
  __bf16*   Zx   = (__bf16*)(ws + OFF_ZX);
  __bf16*   aseq = (__bf16*)(ws + OFF_ASQ);

  // dict order: 0:x 1:emb 2:W_cf 3:W_ff 4:W_uf 5:W_of 6:b_cf 7:b_ff 8:b_uf 9:b_of
  //             10:W_cb 11:W_fb 12:W_ub 13:W_ob 14:b_cb 15:b_fb 16:b_ub 17:b_ob
  //             18:W_p 19:b_p.  Stacked gate order (f,i,o,c) per reference.
  WPtrs p;
  p.W[0] = (const float*)d_in[3];  p.W[1] = (const float*)d_in[4];
  p.W[2] = (const float*)d_in[5];  p.W[3] = (const float*)d_in[2];
  p.W[4] = (const float*)d_in[11]; p.W[5] = (const float*)d_in[12];
  p.W[6] = (const float*)d_in[13]; p.W[7] = (const float*)d_in[10];
  p.b[0] = (const float*)d_in[7];  p.b[1] = (const float*)d_in[8];
  p.b[2] = (const float*)d_in[9];  p.b[3] = (const float*)d_in[6];
  p.b[4] = (const float*)d_in[15]; p.b[5] = (const float*)d_in[16];
  p.b[6] = (const float*)d_in[17]; p.b[7] = (const float*)d_in[14];

  prep_weights_kernel<<<4096, 256, 0, stream>>>(p, Wa, Wx, bg);
  prep_misc_kernel<<<64, 256, 0, stream>>>((const float*)d_in[18], Wp, ab, bar);
  embed_kernel<<<NTB, 128, 0, stream>>>((const int*)d_in[0], (const float*)d_in[1], Xe);
  zx_gemm_kernel<<<dim3(256, 8, 2), 256, 0, stream>>>(Xe, Wx, Zx);
  lstm_recurrent_kernel<<<NWG_REC, 256, 128 * LDSK * 2, stream>>>(Wa, Zx, bg, ab, aseq, bar);
  proj_kernel<<<128, 256, 0, stream>>>(aseq, Wp, (const float*)d_in[19], (float*)d_out);
}
